// SparseMHAEncoder_68118181315087
// MI455X (gfx1250) — matrix operand order repaired and checked by reading, not compiled, not run on hardware
//
#include <hip/hip_runtime.h>
#include <hip/hip_bf16.h>
#include <stdint.h>

typedef __attribute__((ext_vector_type(16))) _Float16 v16h;
typedef __attribute__((ext_vector_type(8)))  _Float16 v8h;
typedef __attribute__((ext_vector_type(4)))  _Float16 v4h;
typedef __attribute__((ext_vector_type(8)))  float    v8f;
typedef __attribute__((ext_vector_type(4)))  float    v4f;

#define LQ_   2048
#define LKV_  8192
#define H_    8
#define B_    4

// ---------------- WMMA helper (CDNA5 16x16x32 f16 -> f32) ------------------

static __device__ inline v8f wmma_f16(v16h a, v16h b, v8f c) {
  return __builtin_amdgcn_wmma_f32_16x16x32_f16(false, a, false, b, (short)0, c,
                                                false, false);
}

// Async global -> LDS copy of 16 bytes (ASYNCcnt-tracked, no VGPR round trip).
static __device__ inline void async_copy_b128(const _Float16* gsrc, _Float16* ldst) {
  uint32_t lds = (uint32_t)(uintptr_t)ldst;     // addr[31:0] == LDS offset
  uint64_t gad = (uint64_t)(uintptr_t)gsrc;
  asm volatile("global_load_async_to_lds_b128 %0, %1, off"
               :: "v"(lds), "v"(gad) : "memory");
}
static __device__ inline void async_wait() {
  asm volatile("s_wait_asynccnt 0x0" ::: "memory");
}

// A fragment: 16x32 (MxK), row-major source with leading dim lda (halves).
static __device__ inline v16h load_a_frag(const _Float16* a, int lda) {
  int lane = threadIdx.x & 31;
  int r  = lane & 15;
  int kb = (lane >> 4) * 16;
  const _Float16* row = a + r * lda;
  v8h x = *(const v8h*)(row + kb);
  v8h y = *(const v8h*)(row + 8 + kb);
  v16h f;
#pragma unroll
  for (int t = 0; t < 8; ++t) { f[t] = x[t]; f[8 + t] = y[t]; }
  return f;
}

// B fragment from an N-major staged tile: bt[n][k], ldk = 32 halves.
static __device__ inline v16h load_b_fragT(const _Float16* bt, int ncol) {
  int lane = threadIdx.x & 31;
  int c  = ncol + (lane & 15);
  int kb = (lane >> 4) * 16;
  return *(const v16h*)(bt + c * 32 + kb);
}

// K^T B-fragment straight from global (rows 64 halves, 128B-aligned).
static __device__ inline v16h load_kT_frag(const _Float16* Kp, int key0, int k0) {
  int lane = threadIdx.x & 31;
  int key  = key0 + (lane & 15);
  int kb   = (lane >> 4) * 16;
  if (key >= 0)
    return *(const v16h*)(Kp + (size_t)key * 64 + k0 + kb);
  v16h z = {};
  return z;
}

// ---------------- Weight transpose: W[512][512] f32 -> Wt[n][k] f16 --------

__global__ __launch_bounds__(256)
void transpose_w(const float* __restrict__ W, _Float16* __restrict__ Wt) {
  __shared__ _Float16 sT[32][33];
  const int tid = threadIdx.x;
  const int k0 = blockIdx.x * 32, n0 = blockIdx.y * 32;
#pragma unroll
  for (int i = 0; i < 4; ++i) {
    int e = tid + 256 * i;
    int r = e >> 5, c = e & 31;
    sT[r][c] = (_Float16)W[(size_t)(k0 + r) * 512 + n0 + c];
  }
  __syncthreads();
#pragma unroll
  for (int i = 0; i < 4; ++i) {
    int e = tid + 256 * i;
    int r = e >> 5, c = e & 31;
    Wt[(size_t)(n0 + r) * 512 + k0 + c] = sT[c][r];
  }
}

// ---------------- Projection GEMM: C[M,512] = A[M,512] @ W[512,512] --------
// Block tile 256x64 (8 waves x 32x64); K-chunk 32; 8 WMMA per wave per step.
// LSH = log2(L) compile-time; VMODE=false: out [b][h][l][d];
// VMODE=true: out [b][h][d][kv] (transposed V for the attention kernel).

template <int LSH, bool VMODE>
__global__ __launch_bounds__(256)
void proj_gemm(const float* __restrict__ A, const _Float16* __restrict__ Wt,
               _Float16* __restrict__ out) {
  __shared__ __align__(32) _Float16 sA[256 * 32];   // row-major [r][k]
  __shared__ __align__(32) _Float16 sBt[64 * 32];   // N-major   [n][k]

  const int tid  = threadIdx.x;
  const int lane = tid & 31;
  const int wv   = tid >> 5;
  const int m0   = blockIdx.x * 256;
  const int n0   = blockIdx.y * 64;

  v8f acc[2][4] = {};

  for (int kk = 0; kk < 512; kk += 32) {
    // B tile (64x32 f16): one async b128 per thread, overlaps A conversion
    {
      int n = tid >> 2, kl = (tid & 3) * 8;
      async_copy_b128(Wt + (size_t)(n0 + n) * 512 + kk + kl, sBt + tid * 8);
    }
    // A tile (256x32 fp32 -> f16): 8 x (float4 load, cvt, v4h store)
#pragma unroll
    for (int i = 0; i < 8; ++i) {
      int elem = (tid + 256 * i) * 4;
      int r = elem >> 5, c = elem & 31;
      v4f fr = *(const v4f*)(A + (size_t)(m0 + r) * 512 + kk + c);
      v4h hh;
#pragma unroll
      for (int t = 0; t < 4; ++t) hh[t] = (_Float16)fr[t];
      *(v4h*)(sA + elem) = hh;
    }
    async_wait();
    __syncthreads();

    v16h a0 = load_a_frag(sA + (wv * 32 + 0)  * 32, 32);
    v16h a1 = load_a_frag(sA + (wv * 32 + 16) * 32, 32);
#pragma unroll
    for (int nt = 0; nt < 4; ++nt) {
      v16h b = load_b_fragT(sBt, nt * 16);
      acc[0][nt] = wmma_f16(a0, b, acc[0][nt]);
      acc[1][nt] = wmma_f16(a1, b, acc[1][nt]);
    }
    __syncthreads();
  }

  constexpr int Lm1 = (1 << LSH) - 1;
#pragma unroll
  for (int s2 = 0; s2 < 2; ++s2) {
#pragma unroll
    for (int nt = 0; nt < 4; ++nt) {
#pragma unroll
      for (int i = 0; i < 8; ++i) {
        int m = m0 + wv * 32 + s2 * 16 + i + 8 * (lane >> 4);
        int n = n0 + nt * 16 + (lane & 15);
        int b = m >> LSH, l = m & Lm1;
        int h = n >> 6, d = n & 63;
        _Float16 val = (_Float16)acc[s2][nt][i];
        if (VMODE)
          out[((((size_t)b * H_ + h) * 64 + d) << LSH) + l] = val;  // V^T layout
        else
          out[((((size_t)b * H_ + h) << LSH) + l) * 64 + d] = val;
      }
    }
  }
}

// ---------------- Output GEMM: d_out[8192,512] = Af16 @ Wo (Wt f16) --------

__global__ __launch_bounds__(256)
void out_gemm(const _Float16* __restrict__ A, const _Float16* __restrict__ Wt,
              float* __restrict__ out) {
  __shared__ __align__(32) _Float16 sA[256 * 32];
  __shared__ __align__(32) _Float16 sBt[64 * 32];

  const int tid  = threadIdx.x;
  const int lane = tid & 31;
  const int wv   = tid >> 5;
  const int m0   = blockIdx.x * 256;
  const int n0   = blockIdx.y * 64;

  v8f acc[2][4] = {};

  for (int kk = 0; kk < 512; kk += 32) {
    {
      int n = tid >> 2, kl = (tid & 3) * 8;
      async_copy_b128(Wt + (size_t)(n0 + n) * 512 + kk + kl, sBt + tid * 8);
    }
#pragma unroll
    for (int i = 0; i < 4; ++i) {                  // A: 256x32 halves, 4/thread
      int idx  = tid + 256 * i;
      int elem = idx * 8;
      int r = elem >> 5, c = elem & 31;
      async_copy_b128(A + (size_t)(m0 + r) * 512 + kk + c, sA + elem);
    }
    async_wait();
    __syncthreads();

    v16h a0 = load_a_frag(sA + (wv * 32 + 0)  * 32, 32);
    v16h a1 = load_a_frag(sA + (wv * 32 + 16) * 32, 32);
#pragma unroll
    for (int nt = 0; nt < 4; ++nt) {
      v16h b = load_b_fragT(sBt, nt * 16);
      acc[0][nt] = wmma_f16(a0, b, acc[0][nt]);
      acc[1][nt] = wmma_f16(a1, b, acc[1][nt]);
    }
    __syncthreads();
  }

#pragma unroll
  for (int s2 = 0; s2 < 2; ++s2) {
#pragma unroll
    for (int nt = 0; nt < 4; ++nt) {
#pragma unroll
      for (int i = 0; i < 8; ++i) {
        int m = m0 + wv * 32 + s2 * 16 + i + 8 * (lane >> 4);
        int n = n0 + nt * 16 + (lane & 15);
        out[(size_t)m * 512 + n] = acc[s2][nt][i];
      }
    }
  }
}

// ---------------- Banded sliding-window attention --------------------------

__global__ __launch_bounds__(32)
void attn_kernel(const _Float16* __restrict__ Qh, const _Float16* __restrict__ Kh,
                 const _Float16* __restrict__ Vt, const float* __restrict__ bias,
                 _Float16* __restrict__ Aout) {
  __shared__ __align__(32) _Float16 sQ[16 * 64];
  __shared__ __align__(32) float    sS[16 * 96];
  __shared__ __align__(32) _Float16 sAttn[16 * 96];
  __shared__ __align__(32) _Float16 sVt[64 * 96];   // [d][j]
  __shared__ float sExtra[16];

  const int lane = threadIdx.x;
  const int bh   = blockIdx.y;
  const int l0   = blockIdx.x * 16;
  const int base = 4 * l0 - 31;

  const _Float16* Qp = Qh + ((size_t)bh * LQ_ + l0) * 64;
  const _Float16* Kp = Kh + (size_t)bh * LKV_ * 64;
  const _Float16* Vp = Vt + (size_t)bh * 64 * LKV_;

#pragma unroll
  for (int i = 0; i < 4; ++i) {
    int e = lane * 8 + i * 256;
    *(v8h*)(sQ + e) = *(const v8h*)(Qp + e);
  }

  // Stage V band transposed: sVt[d][j] = V[base+j][d]
  for (int d = lane >> 4; d < 64; d += 2) {
    int j = (lane & 15);
#pragma unroll
    for (int t = 0; t < 6; ++t) {
      int jj = j + 16 * t;
      int key = base + jj;
      sVt[d * 96 + jj] = (key >= 0) ? Vp[(size_t)d * LKV_ + key] : (_Float16)0.f;
    }
  }
  __syncthreads();

  v16h aq0 = load_a_frag(sQ + 0, 64);
  v16h aq1 = load_a_frag(sQ + 32, 64);

  for (int jt = 0; jt < 6; ++jt) {
    v8f acc = {};
    v16h b0 = load_kT_frag(Kp, base + jt * 16, 0);
    v16h b1 = load_kT_frag(Kp, base + jt * 16, 32);
    acc = wmma_f16(aq0, b0, acc);
    acc = wmma_f16(aq1, b1, acc);
    int col = jt * 16 + (lane & 15);
#pragma unroll
    for (int i = 0; i < 8; ++i) {
      int row = i + 8 * (lane >> 4);
      sS[row * 96 + col] = acc[i];
    }
  }
  __syncthreads();

  if (lane < 16) {
    const int r = lane;
    float sumQ = 0.f;
#pragma unroll
    for (int i = 0; i < 8; ++i) {
      v8h qv = *(const v8h*)(sQ + r * 64 + i * 8);
#pragma unroll
      for (int t = 0; t < 8; ++t) sumQ += (float)qv[t];
    }
    v8h z8 = {};
#pragma unroll
    for (int i = 0; i < 12; ++i) *(v8h*)(sAttn + r * 96 + i * 8) = z8;

    float sc[32];
    float mx = -3.0e38f;
#pragma unroll
    for (int s = 0; s < 32; ++s) {
      int j   = 4 * r + s;
      int kvi = base + j;
      float v = (kvi >= 0) ? 0.125f * (sS[r * 96 + j] + bias[s] * sumQ)
                           : -3.0e38f;
      sc[s] = v;
      mx = fmaxf(mx, v);
    }
    float sum = 0.f;
#pragma unroll
    for (int s = 0; s < 32; ++s) { float e = __expf(sc[s] - mx); sc[s] = e; sum += e; }
    float inv = 1.0f / sum, extra = 0.f;
#pragma unroll
    for (int s = 0; s < 32; ++s) {
      float a = sc[s] * inv;
      sAttn[r * 96 + 4 * r + s] = (_Float16)a;
      extra += a * bias[s];
    }
    sExtra[r] = extra;
  }
  __syncthreads();

  v8f acc2[4] = {};
  for (int ks = 0; ks < 3; ++ks) {
    v16h a = load_a_frag(sAttn + ks * 32, 96);
#pragma unroll
    for (int nt = 0; nt < 4; ++nt) {
      int d  = nt * 16 + (lane & 15);
      int kb = (lane >> 4) * 16;
      v16h b = *(const v16h*)(sVt + d * 96 + ks * 32 + kb);
      acc2[nt] = wmma_f16(a, b, acc2[nt]);
    }
  }

  const int b = bh >> 3, h = bh & 7;
#pragma unroll
  for (int nt = 0; nt < 4; ++nt) {
#pragma unroll
    for (int i = 0; i < 8; ++i) {
      int row = i + 8 * (lane >> 4);
      int d   = nt * 16 + (lane & 15);
      int l   = l0 + row;
      float val = acc2[nt][i] + sExtra[row];
      Aout[((size_t)(b * LQ_ + l)) * 512 + h * 64 + d] = (_Float16)val;
    }
  }
}

// ---------------- Launcher --------------------------------------------------

extern "C" void kernel_launch(void* const* d_in, const int* in_sizes, int n_in,
                              void* d_out, int out_size, void* d_ws, size_t ws_size,
                              hipStream_t stream) {
  (void)in_sizes; (void)n_in; (void)out_size; (void)ws_size;

  const float* q  = (const float*)d_in[0];
  const float* k  = (const float*)d_in[1];
  const float* v  = (const float*)d_in[2];
  const float* Wq = (const float*)d_in[3];
  const float* Wk = (const float*)d_in[4];
  const float* Wv = (const float*)d_in[5];
  const float* Wo = (const float*)d_in[6];
  const float* cb = (const float*)d_in[7];

  _Float16* Qh   = (_Float16*)d_ws;                        // 4*8*2048*64
  _Float16* Kh   = Qh   + (size_t)B_ * H_ * LQ_  * 64;     // 4*8*8192*64
  _Float16* Vth  = Kh   + (size_t)B_ * H_ * LKV_ * 64;     // [b][h][d][kv]
  _Float16* Aout = Vth  + (size_t)B_ * H_ * LKV_ * 64;     // 8192*512
  _Float16* WtQ  = Aout + (size_t)B_ * LQ_ * 512;          // 512*512 each
  _Float16* WtK  = WtQ  + (size_t)512 * 512;
  _Float16* WtV  = WtK  + (size_t)512 * 512;
  _Float16* WtO  = WtV  + (size_t)512 * 512;

  transpose_w<<<dim3(16, 16), 256, 0, stream>>>(Wq, WtQ);
  transpose_w<<<dim3(16, 16), 256, 0, stream>>>(Wk, WtK);
  transpose_w<<<dim3(16, 16), 256, 0, stream>>>(Wv, WtV);
  transpose_w<<<dim3(16, 16), 256, 0, stream>>>(Wo, WtO);

  proj_gemm<11, false><<<dim3((B_ * LQ_) / 256, 8),  256, 0, stream>>>(q, WtQ, Qh);
  proj_gemm<13, false><<<dim3((B_ * LKV_) / 256, 8), 256, 0, stream>>>(k, WtK, Kh);
  proj_gemm<13, true ><<<dim3((B_ * LKV_) / 256, 8), 256, 0, stream>>>(v, WtV, Vth);

  attn_kernel<<<dim3(LQ_ / 16, B_ * H_), 32, 0, stream>>>(Qh, Kh, Vth, cb, Aout);

  out_gemm<<<dim3((B_ * LQ_) / 256, 8), 256, 0, stream>>>(Aout, WtO, (float*)d_out);
}
